// VectorQuantizerEMA_14843406975522
// MI455X (gfx1250) — matrix operand order repaired and checked by reading, not compiled, not run on hardware
//
#include <hip/hip_runtime.h>
#include <hip/hip_bf16.h>
#include <math.h>

// ---------------- problem constants ----------------
#define BATCH 8
#define CCH   64          // channels
#define SDIM  16384       // D*H*W = 16*32*32
#define MCODE 512         // codebook rows
#define NTOT  131072      // BATCH * SDIM
#define TILE_S 512        // spatial positions per workgroup (32 N-subtiles)

// output layout (floats, concatenated in reference return order)
#define Q_OFF    0
#define LOSS_OFF 8388608
#define IDX_OFF  8388609
#define PPL_OFF  8519681
#define USED_OFF 8519682

typedef _Float16 v16h __attribute__((ext_vector_type(16)));
typedef _Float16 v8h  __attribute__((ext_vector_type(8)));
typedef float    v8f  __attribute__((ext_vector_type(8)));
typedef float    v4f  __attribute__((ext_vector_type(4)));

union V16U { v16h v; v8h h[2]; };
union V8FU { v8f v; v4f q[2]; };

// padded row stride (in halves) for LDS tiles: 72*2 = 144 B, multiple of 16 B
#define LPAD 72

// ---------------- kernel 1: codebook prep + accumulator init ----------------
// embf16 stores -2*e (fold the -2 into the GEMM A operand); enorm stores
// |e_f16|^2 computed from the f16-rounded values so distances are >= 0.
__global__ void __launch_bounds__(512)
vq_prep(const float* __restrict__ emb, const float* __restrict__ cluster,
        _Float16* __restrict__ embf16, float* __restrict__ enorm,
        float* __restrict__ counts, float* __restrict__ loss_acc,
        float* __restrict__ out) {
    int m = threadIdx.x;             // 0..511, one code row per thread
    float acc = 0.0f;
    for (int c = 0; c < CCH; ++c) {
        float v = (m == 0) ? 0.0f : ((m == 1) ? 6.0f : emb[m * CCH + c]);
        _Float16 h  = (_Float16)v;           // value used by the WMMA (rounded)
        float    vr = (float)h;
        embf16[m * CCH + c] = (_Float16)(-2.0f * vr);
        acc += vr * vr;
    }
    enorm[m]  = acc;
    counts[m] = 0.0f;
    if (m == 0) *loss_acc = 0.0f;

    __shared__ int sred[512];
    sred[m] = (cluster[m] > 1e-5f) ? 1 : 0;
    __syncthreads();
    for (int s = 256; s > 0; s >>= 1) {
        if (m < s) sred[m] += sred[m + s];
        __syncthreads();
    }
    if (m == 0) out[USED_OFF] = (float)sred[0] / (float)MCODE;
}

// ---------------- kernel 2: WMMA codebook search + quantize ----------------
__device__ __forceinline__ v16h ld_b_frag(const _Float16* lds, int row, int off) {
    const _Float16* base = lds + row * LPAD + off;   // halves off..+7 and off+16..+23
    V16U u;
    u.h[0] = *reinterpret_cast<const v8h*>(base);
    u.h[1] = *reinterpret_cast<const v8h*>(base + 16);
    return u.v;
}
__device__ __forceinline__ v16h ld_a_frag(const _Float16* lds, int row, int c0, int hi) {
    const _Float16* base = lds + row * LPAD;
    V16U u;
    u.h[0] = *reinterpret_cast<const v8h*>(base + c0 + hi * 8);        // K = c0+hi*8 .. +7
    u.h[1] = *reinterpret_cast<const v8h*>(base + c0 + 16 + hi * 8);   // K = c0+16+hi*8 .. +7
    return u.v;
}

__global__ void __launch_bounds__(256)
vq_search(const float* __restrict__ z, const float* __restrict__ embf32,
          const _Float16* __restrict__ embf16, const float* __restrict__ enorm,
          float* __restrict__ counts, float* __restrict__ loss_acc,
          float* __restrict__ out) {
    __shared__ _Float16 ldsA[MCODE * LPAD];   // codebook (-2e), f16, padded rows
    __shared__ _Float16 ldsB[TILE_S * LPAD];  // z tile transposed [spatial][channel]
    __shared__ float    s_en[MCODE];          // |e|^2
    __shared__ float    s_zn[TILE_S];         // |z_f16|^2 per spatial position
    __shared__ int      s_idx[TILE_S];

    const int tid  = threadIdx.x;
    const int lane = tid & 31;
    const int wave = tid >> 5;
    const int hi   = lane >> 4;
    const int lo16 = lane & 15;

    const int b  = blockIdx.x >> 5;          // 32 tiles per batch
    const int s0 = (blockIdx.x & 31) * TILE_S;

    // ---- stage codebook: async global->LDS copy (ASYNCcnt path, no VGPR hop)
    {
        unsigned ldsA_base = (unsigned)(uintptr_t)(void*)&ldsA[0];
        for (int ch = tid; ch < (MCODE * CCH) / 8; ch += 256) {
            int m  = ch >> 3;
            int c8 = (ch & 7) * 8;
            unsigned dst = ldsA_base + (unsigned)(m * LPAD + c8) * 2u;
            unsigned src = (unsigned)(m * CCH + c8) * 2u;
            asm volatile("global_load_async_to_lds_b128 %0, %1, %2"
                         :: "v"(dst), "v"(src), "s"(embf16) : "memory");
        }
    }
    // ---- stage z tile: coalesced f32 reads, transposed f16 writes + |z|^2 ----
    const float* zb = z + (size_t)b * CCH * SDIM + s0;
    {
        float zn0 = 0.0f, zn1 = 0.0f;
        for (int c = 0; c < CCH; ++c) {
            float    zv0 = zb[(size_t)c * SDIM + tid];
            float    zv1 = zb[(size_t)c * SDIM + tid + 256];
            _Float16 h0  = (_Float16)zv0;
            _Float16 h1  = (_Float16)zv1;
            float    r0  = (float)h0;
            float    r1  = (float)h1;
            ldsB[tid * LPAD + c]         = h0;
            ldsB[(tid + 256) * LPAD + c] = h1;
            zn0 += r0 * r0;
            zn1 += r1 * r1;
        }
        s_zn[tid]       = zn0;
        s_zn[tid + 256] = zn1;
    }
    // ---- stage |e|^2 ----
    if (tid < 256) { s_en[tid] = enorm[tid]; s_en[tid + 256] = enorm[tid + 256]; }
    asm volatile("s_wait_asynccnt 0" ::: "memory");
    __syncthreads();

    // ---- WMMA argmin search ----
    // Each wave owns four 16-column N-subtiles; B fragments stay in registers,
    // A fragments + bias are loaded once per code tile and feed 8 WMMAs in
    // four independent accumulator chains (hides WMMA->VALU hazards).
    v16h bA[4], bB[4];
    float zn[4];
    #pragma unroll
    for (int j = 0; j < 4; ++j) {
        const int br = (wave + 8 * j) * 16 + lo16;
        bA[j] = ld_b_frag(ldsB, br, hi * 8);
        bB[j] = ld_b_frag(ldsB, br, 32 + hi * 8);
        zn[j] = s_zn[br] + 1.0f;             // +1 keeps d strictly positive
    }

    unsigned k[4] = { 0xFFFFFFFFu, 0xFFFFFFFFu, 0xFFFFFFFFu, 0xFFFFFFFFu };
    for (int mt = 0; mt < 32; ++mt) {
        const int m0 = mt * 16;
        v16h a0 = ld_a_frag(ldsA, m0 + lo16, 0,  hi);
        v16h a1 = ld_a_frag(ldsA, m0 + lo16, 32, hi);
        const int codeBase = m0 + hi * 8;       // C-layout: vgpr r -> m0 + 8*hi + r
        V8FU en;
        en.q[0] = *reinterpret_cast<const v4f*>(&s_en[codeBase]);
        en.q[1] = *reinterpret_cast<const v4f*>(&s_en[codeBase + 4]);

        v8f acc[4];
        #pragma unroll
        for (int j = 0; j < 4; ++j)             // C init = |e|^2 + |z|^2 + 1
            acc[j] = en.v + zn[j];
        #pragma unroll
        for (int j = 0; j < 4; ++j)
            acc[j] = __builtin_amdgcn_wmma_f32_16x16x32_f16(false, a0, false, bA[j],
                                                            (short)0, acc[j], false, false);
        #pragma unroll
        for (int j = 0; j < 4; ++j)
            acc[j] = __builtin_amdgcn_wmma_f32_16x16x32_f16(false, a1, false, bB[j],
                                                            (short)0, acc[j], false, false);
        // packed-key argmin: d >= 0 so float bits are order-preserving;
        // low 9 bits carry the code for lowest-index tie-break.
        #pragma unroll
        for (int j = 0; j < 4; ++j) {
            #pragma unroll
            for (int r = 0; r < 8; ++r) {
                unsigned c9 = (unsigned)(codeBase + r);
                k[j] = min(k[j], (__float_as_uint(acc[j][r]) & 0xFFFFFE00u) | c9);
            }
        }
    }
    // merge the two half-wave code ranges (same spatial column)
    #pragma unroll
    for (int j = 0; j < 4; ++j)
        k[j] = min(k[j], (unsigned)__shfl_xor((int)k[j], 16));

    if (lane < 16) {
        #pragma unroll
        for (int j = 0; j < 4; ++j) {
            const int sl = (wave + 8 * j) * 16 + lane;
            const int cd = (int)(k[j] & 511u);
            s_idx[sl] = cd;
            out[IDX_OFF + (size_t)b * SDIM + s0 + sl] = (float)cd;
            atomicAdd(&counts[cd], 1.0f);
        }
    }
    __syncthreads();

    // ---- quantize + commitment-loss partial: two spatial positions / thread ----
    {
        float lsum = 0.0f;
        #pragma unroll
        for (int p = 0; p < 2; ++p) {
            const int  t     = tid + p * 256;
            const int  code  = s_idx[t];
            const bool isPin = (code < 2);
            const float pinV = (code == 0) ? 0.0f : 6.0f;
            const float* er  = embf32 + code * CCH;
            const float* zr  = z   + (size_t)b * CCH * SDIM + s0 + t;
            float*       qr  = out + Q_OFF + (size_t)b * CCH * SDIM + s0 + t;
            for (int c = 0; c < CCH; ++c) {
                float q  = isPin ? pinV : er[c];
                float zv = zr[(size_t)c * SDIM];
                qr[(size_t)c * SDIM] = q;
                float d = q - zv;
                lsum += d * d;
            }
        }
        #pragma unroll
        for (int off = 16; off > 0; off >>= 1) lsum += __shfl_xor(lsum, off);
        if (lane == 0) atomicAdd(loss_acc, lsum);
    }
}

// ---------------- kernel 3: finalize perplexity + loss ----------------
__global__ void __launch_bounds__(512)
vq_finalize(const float* __restrict__ counts, const float* __restrict__ loss_acc,
            float* __restrict__ out) {
    int m = threadIdx.x;
    __shared__ float sred[512];
    float p = counts[m] / (float)NTOT;
    sred[m] = p * logf(p + 1e-10f);
    __syncthreads();
    for (int s = 256; s > 0; s >>= 1) {
        if (m < s) sred[m] += sred[m + s];
        __syncthreads();
    }
    if (m == 0) {
        out[PPL_OFF]  = expf(-sred[0]);
        out[LOSS_OFF] = 0.25f * (*loss_acc) / (float)((size_t)NTOT * CCH);
    }
}

// ---------------- launcher ----------------
extern "C" void kernel_launch(void* const* d_in, const int* in_sizes, int n_in,
                              void* d_out, int out_size, void* d_ws, size_t ws_size,
                              hipStream_t stream) {
    const float* z       = (const float*)d_in[0];
    const float* emb     = (const float*)d_in[1];
    const float* cluster = (const float*)d_in[2];
    float* out = (float*)d_out;

    // workspace carve-up (hipMalloc'd => 256B aligned)
    char* w = (char*)d_ws;
    _Float16* embf16 = (_Float16*)(w);                 // 512*64*2   = 65536 B
    float*    enorm  = (float*)(w + 65536);            // 512*4      =  2048 B
    float*    counts = (float*)(w + 65536 + 2048);     // 512*4      =  2048 B
    float*    lossA  = (float*)(w + 65536 + 4096);     // 4 B

    vq_prep<<<1, 512, 0, stream>>>(emb, cluster, embf16, enorm, counts, lossA, out);
    vq_search<<<BATCH * (SDIM / TILE_S), 256, 0, stream>>>(z, emb, embf16, enorm,
                                                           counts, lossA, out);
    vq_finalize<<<1, 512, 0, stream>>>(counts, lossA, out);
}
